// GeneralAttention_43679817400719
// MI455X (gfx1250) — matrix operand order repaired and checked by reading, not compiled, not run on hardware
//
#include <hip/hip_runtime.h>

// ---------------------------------------------------------------------------
// GeneralAttention for MI455X (gfx1250, wave32, WMMA bf16 16x16x32)
//   score, ctx = softmax((query·Wt)·keysT + mask) , score·values
// All GEMMs run on v_wmma_f32_16x16x32_bf16 with f32 accumulation.
// K loop: 2x-unrolled ping-pong (no register copies, loads of half i+1 in
// flight during WMMAs of half i). All hot-path pointers are pinned to the
// global address space so every access lowers to global_load/global_store
// (LOADcnt/STOREcnt only, never flat_* which would couple to DScnt).
// ---------------------------------------------------------------------------

#define TQ 1024
#define NB 16
#define QD 1024
#define TK 1024
#define KD 1024
#define VD 1024

#define GLB __attribute__((address_space(1)))

typedef __attribute__((ext_vector_type(16))) __bf16 bf16x16;
typedef __attribute__((ext_vector_type(8)))  __bf16 bf16x8;
typedef __attribute__((ext_vector_type(4)))  __bf16 bf16x4;
typedef __attribute__((ext_vector_type(8)))  float  f32x8;
typedef __attribute__((ext_vector_type(4)))  float  f32x4;

// ---- WMMA wrapper ----------------------------------------------------------
static __device__ inline f32x8 wmma_bf16(bf16x16 a, bf16x16 b, f32x8 c) {
  return __builtin_amdgcn_wmma_f32_16x16x32_bf16(
      /*neg_a=*/false, a, /*neg_b=*/false, b,
      /*c_mod=*/(short)0, c, /*reuse_a=*/false, /*reuse_b=*/false);
}

// ---- Operand loaders (contiguous k = 16*(L>>4)+s in both A and B, wave32) -------------
// A (16x32 bf16): lane L holds row (L&15); K chunk = {kb..kb+15},
// kb = 16*(L>>4). Two 16-byte global loads per lane.
static __device__ inline bf16x16 load_a_bf16(const GLB __bf16* row, int lane) {
  const int kb = (lane >> 4) << 4;
  bf16x8 lo = *(const GLB bf16x8*)(row + kb);
  bf16x8 hi = *(const GLB bf16x8*)(row + kb + 8);
  bf16x16 a;
#pragma unroll
  for (int i = 0; i < 8; ++i) { a[i] = lo[i]; a[i + 8] = hi[i]; }
  return a;
}

// A operand built from f32 memory (score rows), converted to bf16 in-register.
static __device__ inline bf16x16 load_a_f32(const GLB float* row, int lane) {
  const int kb = (lane >> 4) << 4;
  f32x4 a0 = *(const GLB f32x4*)(row + kb);
  f32x4 a1 = *(const GLB f32x4*)(row + kb + 4);
  f32x4 a2 = *(const GLB f32x4*)(row + kb + 8);
  f32x4 a3 = *(const GLB f32x4*)(row + kb + 12);
  bf16x16 a;
#pragma unroll
  for (int i = 0; i < 4; ++i) {
    a[i]      = (__bf16)a0[i];
    a[i + 4]  = (__bf16)a1[i];
    a[i + 8]  = (__bf16)a2[i];
    a[i + 12] = (__bf16)a3[i];
  }
  return a;
}

// B (32x16 bf16): lane n holds column (n&15); K chunk = 16 contiguous elems
// starting at 16*(n>>4): one 32-byte global load per lane.
static __device__ inline bf16x16 load_b_vec(const GLB __bf16* p) {
  return *(const GLB bf16x16*)p;
}

// ---- Prep: dtype conversion / layout kernels -------------------------------

// query [t][b][q] f32 -> Qb [b][t][q] bf16   (both sides coalesced over q)
__global__ __launch_bounds__(256) void convert_query_kernel(
    const float* __restrict__ q, __bf16* __restrict__ Qb) {
  size_t i4 = (size_t)blockIdx.x * blockDim.x + threadIdx.x; // vec4 index
  int q4  = (int)(i4 % (QD / 4));
  size_t r = i4 / (QD / 4);
  int t = (int)(r % TQ);
  int b = (int)(r / TQ);
  f32x4 v = *(const f32x4*)(q + ((size_t)t * NB + b) * QD + (size_t)q4 * 4);
  bf16x4 o;
#pragma unroll
  for (int i = 0; i < 4; ++i) o[i] = (__bf16)v[i];
  *(bf16x4*)(Qb + i4 * 4) = o;
}

// straight f32 -> bf16 cast (W, keys)
__global__ __launch_bounds__(256) void cast_bf16_kernel(
    const float* __restrict__ in, __bf16* __restrict__ out, int n4) {
  int i = blockIdx.x * blockDim.x + threadIdx.x;
  if (i >= n4) return;
  f32x4 v = *(const f32x4*)(in + (size_t)i * 4);
  bf16x4 o;
#pragma unroll
  for (int k = 0; k < 4; ++k) o[k] = (__bf16)v[k];
  *(bf16x4*)(out + (size_t)i * 4) = o;
}

// values [b][s][v] f32 -> Vt [b][v][s] bf16 via 32x32 LDS tile transpose
__global__ __launch_bounds__(256) void transpose_values_kernel(
    const float* __restrict__ V, __bf16* __restrict__ Vt) {
  __shared__ float tile[32][33];
  const int b  = blockIdx.z;
  const int s0 = blockIdx.x * 32;
  const int v0 = blockIdx.y * 32;
  const int tx = threadIdx.x;           // 0..31
  const int ty = threadIdx.y;           // 0..7
#pragma unroll
  for (int i = 0; i < 4; ++i) {
    int s = ty + i * 8;
    tile[s][tx] = V[((size_t)b * TK + s0 + s) * VD + v0 + tx];
  }
  __syncthreads();
#pragma unroll
  for (int i = 0; i < 4; ++i) {
    int v = ty + i * 8;
    Vt[((size_t)b * VD + v0 + v) * TK + s0 + tx] = (__bf16)tile[tx][v];
  }
}

// ---- Ping-pong GEMM body (A bf16), 4 N-tiles per wave ----------------------
static __device__ inline void gemm_k_loop_bf16(const GLB __bf16* ap,
                                               const GLB __bf16* bq0,
                                               const GLB __bf16* bq1,
                                               const GLB __bf16* bq2,
                                               const GLB __bf16* bq3,
                                               const void* pf, int kdim,
                                               int lane, f32x8 acc[4]) {
  const GLB __bf16* bp[4] = {bq0, bq1, bq2, bq3};
  const char* pfp = (const char*)pf;
  bf16x16 a0 = load_a_bf16(ap, lane);
  bf16x16 b0[4];
#pragma unroll
  for (int nt = 0; nt < 4; ++nt) b0[nt] = load_b_vec(bp[nt]);

  for (int k0 = 0; k0 < kdim - 64; k0 += 64) {
    bf16x16 a1 = load_a_bf16(ap + 32, lane);
    bf16x16 b1[4];
#pragma unroll
    for (int nt = 0; nt < 4; ++nt) b1[nt] = load_b_vec(bp[nt] + 32);
#pragma unroll
    for (int nt = 0; nt < 4; ++nt) acc[nt] = wmma_bf16(a0, b0[nt], acc[nt]);

    a0 = load_a_bf16(ap + 64, lane);
#pragma unroll
    for (int nt = 0; nt < 4; ++nt) b0[nt] = load_b_vec(bp[nt] + 64);
#pragma unroll
    for (int nt = 0; nt < 4; ++nt) acc[nt] = wmma_bf16(a1, b1[nt], acc[nt]);

    ap += 64;
#pragma unroll
    for (int nt = 0; nt < 4; ++nt) bp[nt] += 64;
    __builtin_prefetch(pfp + 640, 0, 3);   // A stream, ~10 K-steps ahead
    pfp += 128;
  }
  // tail: last two K-steps
  bf16x16 a1 = load_a_bf16(ap + 32, lane);
  bf16x16 b1[4];
#pragma unroll
  for (int nt = 0; nt < 4; ++nt) b1[nt] = load_b_vec(bp[nt] + 32);
#pragma unroll
  for (int nt = 0; nt < 4; ++nt) acc[nt] = wmma_bf16(a0, b0[nt], acc[nt]);
#pragma unroll
  for (int nt = 0; nt < 4; ++nt) acc[nt] = wmma_bf16(a1, b1[nt], acc[nt]);
}

// ---- GEMM 1: QP[m][k] = Qb[m][:] · Wb[k][:]   (m = b*TQ + t) ---------------
__global__ __launch_bounds__(128) void proj_kernel(
    const __bf16* __restrict__ Qb, const __bf16* __restrict__ Wb,
    __bf16* __restrict__ QP) {
  const int lane = threadIdx.x & 31;
  const int wave = threadIdx.x >> 5;
  const int m0 = blockIdx.y * 16;
  const int n0 = blockIdx.x * 256 + wave * 64;

  const __bf16* ap_g = Qb + (size_t)(m0 + (lane & 15)) * QD;
  const GLB __bf16* ap = (const GLB __bf16*)ap_g;
  const int bk = (lane >> 4) << 4;
  const GLB __bf16* bp[4];
#pragma unroll
  for (int nt = 0; nt < 4; ++nt)
    bp[nt] =
        (const GLB __bf16*)(Wb + (size_t)(n0 + nt * 16 + (lane & 15)) * QD) +
        bk;

  f32x8 acc[4] = {};
  gemm_k_loop_bf16(ap, bp[0], bp[1], bp[2], bp[3], ap_g, QD, lane, acc);

  // D layout: VGPR r / lanes 0-15 -> (M=r, N=lane); lanes 16-31 -> (M=r+8).
  GLB __bf16* qp = (GLB __bf16*)QP;
  const int col = lane & 15;
  const int rh  = (lane >> 4) * 8;
#pragma unroll
  for (int nt = 0; nt < 4; ++nt)
#pragma unroll
    for (int r = 0; r < 8; ++r)
      qp[(size_t)(m0 + rh + r) * KD + n0 + nt * 16 + col] = (__bf16)acc[nt][r];
}

// ---- GEMM 2 + mask + fused softmax -----------------------------------------
// One block (16 waves, 512 thr) per (b, 16-query strip). Each wave covers 64
// key columns (4 accumulators), ping-pong K loop. 16x1024 f32 slab in LDS,
// then full-wave row softmax, normalized f32 score -> d_out.
#define SSTRIDE 1024
__global__ __launch_bounds__(512) void scores_kernel(
    const __bf16* __restrict__ QP, const __bf16* __restrict__ Kb,
    const float* __restrict__ mask, float* __restrict__ score) {
  __shared__ float S[16 * SSTRIDE];   // 64 KB
  const int lane = threadIdx.x & 31;
  const int wave = threadIdx.x >> 5;
  const int b  = blockIdx.y;
  const int t0 = blockIdx.x * 16;
  const int sw = wave * 64;           // this wave's key-column base

  const __bf16* ap_g = QP + (size_t)(b * TQ + t0 + (lane & 15)) * KD;
  const GLB __bf16* ap = (const GLB __bf16*)ap_g;
  const __bf16* kbase = Kb + (size_t)b * TK * KD;
  const int bk = (lane >> 4) << 4;
  const GLB __bf16* bp[4];
#pragma unroll
  for (int nt = 0; nt < 4; ++nt)
    bp[nt] = (const GLB __bf16*)(kbase +
                                 (size_t)(sw + nt * 16 + (lane & 15)) * KD) +
             bk;

  f32x8 acc[4] = {};
  gemm_k_loop_bf16(ap, bp[0], bp[1], bp[2], bp[3], ap_g, KD, lane, acc);

  const GLB float* gmask = (const GLB float*)mask;
  const int col = lane & 15;
  const int rh  = (lane >> 4) * 8;
#pragma unroll
  for (int nt = 0; nt < 4; ++nt) {
    float mval = gmask[b * TK + sw + nt * 16 + col];  // additive, per column
#pragma unroll
    for (int r = 0; r < 8; ++r)
      S[(rh + r) * SSTRIDE + sw + nt * 16 + col] = acc[nt][r] + mval;
  }
  __syncthreads();

  // softmax: one full wave (32 lanes) per row
  const int row = threadIdx.x >> 5;   // == wave, 0..15
  float* srow = &S[row * SSTRIDE];

  float m = -3.0e38f;
  for (int c = lane; c < TK; c += 32) m = fmaxf(m, srow[c]);
  m = fmaxf(m, __shfl_xor(m, 1));
  m = fmaxf(m, __shfl_xor(m, 2));
  m = fmaxf(m, __shfl_xor(m, 4));
  m = fmaxf(m, __shfl_xor(m, 8));
  m = fmaxf(m, __shfl_xor(m, 16));

  float sum = 0.0f;
  for (int c = lane; c < TK; c += 32) {
    float e = __expf(srow[c] - m);
    srow[c] = e;
    sum += e;
  }
  sum += __shfl_xor(sum, 1);
  sum += __shfl_xor(sum, 2);
  sum += __shfl_xor(sum, 4);
  sum += __shfl_xor(sum, 8);
  sum += __shfl_xor(sum, 16);
  const float inv = 1.0f / sum;

  GLB float* orow =
      (GLB float*)(score + (size_t)(b * TQ + t0 + row) * TK);
  for (int c = lane; c < TK; c += 32) orow[c] = srow[c] * inv;
}

// ---- GEMM 3: ctx[b][t][v] = score[b][t][:] · Vt[b][v][:] -------------------
__global__ __launch_bounds__(128) void ctx_kernel(
    const float* __restrict__ score, const __bf16* __restrict__ Vt,
    float* __restrict__ ctx) {
  const int lane = threadIdx.x & 31;
  const int wave = threadIdx.x >> 5;
  const int b  = blockIdx.z;
  const int t0 = blockIdx.y * 16;
  const int v0 = blockIdx.x * 256 + wave * 64;

  const float* ap_g = score + (size_t)(b * TQ + t0 + (lane & 15)) * TK;
  const GLB float* ap = (const GLB float*)ap_g;
  const __bf16* vbase = Vt + (size_t)b * VD * TK;
  const int bk = (lane >> 4) << 4;
  const GLB __bf16* bp[4];
#pragma unroll
  for (int nt = 0; nt < 4; ++nt)
    bp[nt] = (const GLB __bf16*)(vbase +
                                 (size_t)(v0 + nt * 16 + (lane & 15)) * TK) +
             bk;

  f32x8 acc[4] = {};
  const char* pfp = (const char*)ap_g;
  // ping-pong K loop, A converted from f32 score rows in-register
  bf16x16 a0 = load_a_f32(ap, lane);
  bf16x16 b0[4];
#pragma unroll
  for (int nt = 0; nt < 4; ++nt) b0[nt] = load_b_vec(bp[nt]);

  for (int s0 = 0; s0 < TK - 64; s0 += 64) {
    bf16x16 a1 = load_a_f32(ap + 32, lane);
    bf16x16 b1[4];
#pragma unroll
    for (int nt = 0; nt < 4; ++nt) b1[nt] = load_b_vec(bp[nt] + 32);
#pragma unroll
    for (int nt = 0; nt < 4; ++nt) acc[nt] = wmma_bf16(a0, b0[nt], acc[nt]);

    a0 = load_a_f32(ap + 64, lane);
#pragma unroll
    for (int nt = 0; nt < 4; ++nt) b0[nt] = load_b_vec(bp[nt] + 64);
#pragma unroll
    for (int nt = 0; nt < 4; ++nt) acc[nt] = wmma_bf16(a1, b1[nt], acc[nt]);

    ap += 64;
#pragma unroll
    for (int nt = 0; nt < 4; ++nt) bp[nt] += 64;
    __builtin_prefetch(pfp + 1280, 0, 3);
    pfp += 256;
  }
  {
    bf16x16 a1 = load_a_f32(ap + 32, lane);
    bf16x16 b1[4];
#pragma unroll
    for (int nt = 0; nt < 4; ++nt) b1[nt] = load_b_vec(bp[nt] + 32);
#pragma unroll
    for (int nt = 0; nt < 4; ++nt) acc[nt] = wmma_bf16(a0, b0[nt], acc[nt]);
#pragma unroll
    for (int nt = 0; nt < 4; ++nt) acc[nt] = wmma_bf16(a1, b1[nt], acc[nt]);
  }

  GLB float* octx = (GLB float*)ctx;
  const int col = lane & 15;
  const int rh  = (lane >> 4) * 8;
#pragma unroll
  for (int nt = 0; nt < 4; ++nt)
#pragma unroll
    for (int r = 0; r < 8; ++r)
      octx[(size_t)(b * TQ + t0 + rh + r) * VD + v0 + nt * 16 + col] =
          acc[nt][r];
}

// ---------------------------------------------------------------------------
extern "C" void kernel_launch(void* const* d_in, const int* in_sizes, int n_in,
                              void* d_out, int out_size, void* d_ws,
                              size_t ws_size, hipStream_t stream) {
  const float* query  = (const float*)d_in[0]; // [TQ][NB][QD]
  const float* keys   = (const float*)d_in[1]; // [NB][TK][KD]
  const float* values = (const float*)d_in[2]; // [NB][TK][VD]
  const float* mask   = (const float*)d_in[3]; // [NB][TK]
  const float* W      = (const float*)d_in[4]; // [KD][QD]

  float* score = (float*)d_out;                        // [NB][TQ][TK]
  float* ctx   = score + (size_t)NB * TQ * TK;         // [NB][TQ][VD]

  // workspace layout (bf16):  Qb | Kb | Vt | Wb | QP   (~130 MiB total)
  __bf16* Qb = (__bf16*)d_ws;                          // [NB][TQ][QD]
  __bf16* Kb = Qb + (size_t)NB * TQ * QD;              // [NB][TK][KD]
  __bf16* Vt = Kb + (size_t)NB * TK * KD;              // [NB][VD][TK]
  __bf16* Wb = Vt + (size_t)NB * VD * TK;              // [KD][QD]
  __bf16* QP = Wb + (size_t)KD * QD;                   // [NB*TQ][KD]

  // --- prep: conversions / transposes ---
  {
    int n4 = NB * TQ * QD / 4;
    convert_query_kernel<<<n4 / 256, 256, 0, stream>>>(query, Qb);
  }
  {
    int n4 = KD * QD / 4;
    cast_bf16_kernel<<<(n4 + 255) / 256, 256, 0, stream>>>(W, Wb, n4);
  }
  {
    int n4 = NB * TK * KD / 4;
    cast_bf16_kernel<<<(n4 + 255) / 256, 256, 0, stream>>>(keys, Kb, n4);
  }
  transpose_values_kernel<<<dim3(TK / 32, VD / 32, NB), dim3(32, 8), 0,
                            stream>>>(values, Vt);

  // --- GEMM 1: query projection (M=NB*TQ, N=KD, K=QD) ---
  proj_kernel<<<dim3(KD / 256, NB * TQ / 16), 128, 0, stream>>>(Qb, Wb, QP);

  // --- GEMM 2 + mask + softmax ---
  scores_kernel<<<dim3(TQ / 16, NB), 512, 0, stream>>>(QP, Kb, mask, score);

  // --- GEMM 3: context ---
  ctx_kernel<<<dim3(VD / 256, TQ / 16, NB), 128, 0, stream>>>(score, Vt, ctx);
}
